// WeightNet_70703751627297
// MI455X (gfx1250) — matrix operand order repaired and checked by reading, not compiled, not run on hardware
//
#include <hip/hip_runtime.h>
#include <math.h>

// ---------- types for WMMA ----------
typedef __attribute__((ext_vector_type(16))) __bf16 v16bf;
typedef __attribute__((ext_vector_type(8)))  __bf16 v8bf;
typedef __attribute__((ext_vector_type(8)))  float  v8f;

// ---------- problem constants ----------
#define INPC   256
#define OUPC   256
#define HDIM   28
#define WDIM   28
#define NSP    784              // 28*28
#define BM     128              // M tile per block (output channels)
#define BN     112              // N tile per block (4 image rows)
#define KC     32               // K chunk (input channels per step)
#define NBATCH 32

// workspace layout (fast path)
#define A_BYTES  (32 * 512 * 4)                       // sigmoid activations
#define X_BYTES  (32 * 256 * 784 * 4)                 // packed bf16 hi|lo x
#define W_BYTES  ((size_t)32 * 9 * 256 * 256 * 2)     // one bf16 weight plane
#define WS_NEED  ((size_t)A_BYTES + X_BYTES + 2 * W_BYTES)

__device__ __forceinline__ unsigned short f2bf(float v) {   // RNE float->bf16
    unsigned u = __float_as_uint(v);
    return (unsigned short)((u + 0x7FFFu + ((u >> 16) & 1u)) >> 16);
}
__device__ __forceinline__ float bf2f(unsigned short h) {
    return __uint_as_float(((unsigned)h) << 16);
}

// ===================================================================
// Kernel 1: a[b,g] = sigmoid(x_gap[b,:] . w1[g,:] + b1[g]),  32x512
// ===================================================================
__global__ void wn_fc1(const float* __restrict__ x_gap,
                       const float* __restrict__ w1,
                       const float* __restrict__ b1,
                       float* __restrict__ a_out) {
    int gid = blockIdx.x * blockDim.x + threadIdx.x;   // 0..16383
    int b = gid >> 9;
    int g = gid & 511;
    const float* xg = x_gap + b * 16;
    const float* w  = w1 + g * 16;
    float s = b1[g];
#pragma unroll
    for (int c = 0; c < 16; ++c) s += xg[c] * w[c];
    a_out[gid] = 1.0f / (1.0f + __expf(-s));
}

// ===================================================================
// Kernel 1b (fast path): pack x as bf16 (hi|lo) u32, same [b][c][h][w]
// ===================================================================
__global__ void wn_xgen(const float* __restrict__ x, unsigned* __restrict__ xpre) {
    int gid = blockIdx.x * blockDim.x + threadIdx.x;   // 6,422,528 total
    float v = x[gid];
    unsigned short h = f2bf(v);
    unsigned short l = f2bf(v - bf2f(h));
    xpre[gid] = ((unsigned)l << 16) | (unsigned)h;
}

// ===================================================================
// Kernel 1c (fast path): scaled dynamic weights, split bf16 hi/lo
// planes, layout [b][tap][o][i] with i contiguous (matches LDS tile).
// ===================================================================
__global__ void wn_wgen(const float* __restrict__ w2,
                        const float* __restrict__ a_vec,
                        unsigned short* __restrict__ wh,
                        unsigned short* __restrict__ wl) {
    int gid  = blockIdx.x * blockDim.x + threadIdx.x;  // 18,874,368 total
    int i    = gid & 255;
    int o    = (gid >> 8) & 255;
    int rest = gid >> 16;                              // b*9 + t
    int t    = rest % 9;
    int b    = rest / 9;
    float v = a_vec[b * 512 + 2 * o + (i >= 128 ? 1 : 0)] *
              w2[(size_t)o * 2304 + i * 9 + t];
    unsigned short h = f2bf(v);
    unsigned short l = f2bf(v - bf2f(h));
    wh[gid] = h;
    wl[gid] = l;
}

// ===================================================================
// Kernel 2: per-sample implicit-GEMM 3x3 conv on the WMMA pipe.
//   grid = (7, 2, 32); block = 256 threads = 8 waves (wave32)
//   wave w owns M rows [16w,16w+16) x 7 N-subtiles of 16
// Precision: bf16 hi/lo split, acc += Ah*Bh + Ah*Bl + Al*Bh (fp32 acc)
// FAST: weight tiles arrive via global_load_async_to_lds_b128,
//       double-buffered and overlapped with the previous tap's WMMAs.
// ===================================================================
template <bool FAST>
__global__ void __launch_bounds__(256)
wn_conv(const float* __restrict__ x,
        const unsigned* __restrict__ xpre,
        const float* __restrict__ w2,
        const unsigned short* __restrict__ wpre_hi,
        const unsigned short* __restrict__ wpre_lo,
        const float* __restrict__ a_vec,
        float* __restrict__ out) {
    // x slab with halo, layout [row 0..5][col 0..31][ch 0..31]
    __shared__ unsigned short slab_hi[6 * 32 * 32];
    __shared__ unsigned short slab_lo[6 * 32 * 32];
    // weight tile for one tap, layout [m 0..127][k 0..31]; double-buffered
    __shared__ unsigned short wa_hi[2][BM * KC];
    __shared__ unsigned short wa_lo[2][BM * KC];

    const int tid   = threadIdx.x;
    const int lane  = tid & 31;
    const int wave  = tid >> 5;
    const int b     = blockIdx.z;
    const int obase = blockIdx.y * BM;
    const int nbase = blockIdx.x * BN;
    const int h0    = blockIdx.x * 4;          // first image row of this tile

    const float*    xb  = x    + (size_t)b * INPC * NSP;
    const unsigned* xpb = xpre + (size_t)b * INPC * NSP;
    const float*    av  = a_vec + b * 512;

    v8f acc[7] = {};                           // 7 N-subtiles, fp32 C/D

    // B fragment (contiguous K: lanes 0-15 -> K 0..15, lanes 16-31 -> K 16..31)
    const int khalf = (lane < 16) ? 0 : 16;
    const int nloc  = lane & 15;
    // A fragment (contiguous K: lanes 0-15 -> K 0..15, lanes 16-31 -> K 16..31)
    const int mA  = (lane & 15) + wave * 16;
    const int kA0 = (lane < 16) ? 0 : 16;
    const int kA1 = kA0 + 8;

    // fast path: async tile copy helper state (per-thread 16B units)
    const int cp_m   = tid >> 2;               // 2 rows per 8 threads... m = unit/4
    const int cp_seg = tid & 3;                // 16B segment within 64B row
    const unsigned lds_hi0 = (unsigned)(size_t)&wa_hi[0][0];
    const unsigned lds_lo0 = (unsigned)(size_t)&wa_lo[0][0];

    for (int ic = 0; ic < 8; ++ic) {
        const int icbase = ic * KC;
        __syncthreads();                       // prior reads of slab/wa done
        // ---- load x slab (6 rows x 30 cols x 32 ch), zero-padded halo ----
        for (int idx = tid; idx < 6 * 30 * 32; idx += 256) {
            int cc   = idx % 30;               // col fastest -> coalesced global
            int rest = idx / 30;
            int rr   = rest % 6;
            int ch   = rest / 6;
            int hg   = h0 + rr - 1;
            int wg   = cc - 1;
            int off  = (rr * 32 + cc) * 32 + ch;
            if (FAST) {
                unsigned p = 0;
                if (hg >= 0 && hg < HDIM && wg >= 0 && wg < WDIM)
                    p = xpb[(size_t)(icbase + ch) * NSP + hg * WDIM + wg];
                slab_hi[off] = (unsigned short)(p & 0xffffu);
                slab_lo[off] = (unsigned short)(p >> 16);
            } else {
                float v = 0.0f;
                if (hg >= 0 && hg < HDIM && wg >= 0 && wg < WDIM)
                    v = xb[(size_t)(icbase + ch) * NSP + hg * WDIM + wg];
                unsigned short h = f2bf(v);
                slab_hi[off] = h;
                slab_lo[off] = f2bf(v - bf2f(h));
            }
        }
        if (FAST) {
            // issue async copy of tap-0 weight tile into buffer 0
            const size_t rowbase = ((size_t)(b * 9 + 0) * 256 + obase + cp_m) * 256;
            const unsigned goff = (unsigned)(icbase * 2 + cp_seg * 16);
            const char* ghi = (const char*)(wpre_hi + rowbase) + goff;
            const char* glo = (const char*)(wpre_lo + rowbase) + goff;
            unsigned ldst = (unsigned)(cp_m * 64 + cp_seg * 16);
#pragma unroll
            for (int half = 0; half < 2; ++half) {   // rows [0..63] then [64..127]
                asm volatile("global_load_async_to_lds_b128 %0, %1, off"
                             :: "v"(lds_hi0 + ldst + half * 64 * 64),
                                "v"(ghi + (size_t)half * 64 * 512) : "memory");
                asm volatile("global_load_async_to_lds_b128 %0, %1, off"
                             :: "v"(lds_lo0 + ldst + half * 64 * 64),
                                "v"(glo + (size_t)half * 64 * 512) : "memory");
            }
        }
        for (int t = 0; t < 9; ++t) {
            const int dy = t / 3, dx = t % 3;
            int buf;
            if (FAST) {
                buf = t & 1;
                asm volatile("s_wait_asynccnt 0x0" ::: "memory");
                __syncthreads();               // tile t + slab visible everywhere
                if (t < 8) {                   // overlap tap t+1 copy with WMMAs
                    const int tn = t + 1;
                    const size_t rowbase =
                        ((size_t)(b * 9 + tn) * 256 + obase + cp_m) * 256;
                    const unsigned goff = (unsigned)(icbase * 2 + cp_seg * 16);
                    const char* ghi = (const char*)(wpre_hi + rowbase) + goff;
                    const char* glo = (const char*)(wpre_lo + rowbase) + goff;
                    unsigned ldst = (unsigned)(cp_m * 64 + cp_seg * 16) +
                                    (tn & 1) * (unsigned)(BM * KC * 2);
#pragma unroll
                    for (int half = 0; half < 2; ++half) {
                        asm volatile("global_load_async_to_lds_b128 %0, %1, off"
                                     :: "v"(lds_hi0 + ldst + half * 64 * 64),
                                        "v"(ghi + (size_t)half * 64 * 512) : "memory");
                        asm volatile("global_load_async_to_lds_b128 %0, %1, off"
                                     :: "v"(lds_lo0 + ldst + half * 64 * 64),
                                        "v"(glo + (size_t)half * 64 * 512) : "memory");
                    }
                }
            } else {
                buf = 0;
                __syncthreads();               // slab visible / prior wa reads done
                for (int idx = tid; idx < BM * KC; idx += 256) {
                    int k = idx & 31;
                    int m = idx >> 5;
                    float s = av[2 * (obase + m) + ((icbase + k) >= 128 ? 1 : 0)];
                    float v = w2[(size_t)(obase + m) * 2304 + (icbase + k) * 9 + t] * s;
                    unsigned short h = f2bf(v);
                    wa_hi[0][idx] = h;
                    wa_lo[0][idx] = f2bf(v - bf2f(h));
                }
                __syncthreads();               // wa tile visible
            }
            // ---- A fragments (two 16B chunks per lane, hi & lo) ----
            const unsigned short* wh = &wa_hi[buf][0];
            const unsigned short* wl = &wa_lo[buf][0];
            union { v16bf v; v8bf h[2]; } ahf, alf;
            ahf.h[0] = *(const v8bf*)&wh[mA * KC + kA0];
            ahf.h[1] = *(const v8bf*)&wh[mA * KC + kA1];
            alf.h[0] = *(const v8bf*)&wl[mA * KC + kA0];
            alf.h[1] = *(const v8bf*)&wl[mA * KC + kA1];
#pragma unroll
            for (int j = 0; j < 7; ++j) {
                int nl   = j * 16 + nloc;          // 0..111 within tile
                int srow = nl / 28 + dy;           // shifted slab row
                int scol = nl % 28 + dx;           // shifted slab col
                int soff = (srow * 32 + scol) * 32 + khalf;   // 32B aligned
                v16bf bh = *(const v16bf*)&slab_hi[soff];
                v16bf bl = *(const v16bf*)&slab_lo[soff];
                acc[j] = __builtin_amdgcn_wmma_f32_16x16x32_bf16(
                    false, ahf.v, false, bh, (short)0, acc[j], false, false);
                acc[j] = __builtin_amdgcn_wmma_f32_16x16x32_bf16(
                    false, ahf.v, false, bl, (short)0, acc[j], false, false);
                acc[j] = __builtin_amdgcn_wmma_f32_16x16x32_bf16(
                    false, alf.v, false, bh, (short)0, acc[j], false, false);
            }
        }
    }

    // ---- epilogue: C/D layout VGPR r -> M=r (lanes 0-15) / M=r+8 (16-31) ----
    const int orow = wave * 16 + (lane < 16 ? 0 : 8);
    float* ob = out + ((size_t)b * OUPC + obase) * NSP;
#pragma unroll
    for (int j = 0; j < 7; ++j) {
        int n = nbase + j * 16 + nloc;
#pragma unroll
        for (int r = 0; r < 8; ++r) {
            ob[(size_t)(orow + r) * NSP + n] = acc[j][r];
        }
    }
}

// ===================================================================
extern "C" void kernel_launch(void* const* d_in, const int* in_sizes, int n_in,
                              void* d_out, int out_size, void* d_ws, size_t ws_size,
                              hipStream_t stream) {
    const float* x     = (const float*)d_in[0];  // [32,256,28,28]
    const float* x_gap = (const float*)d_in[1];  // [32,16,1,1]
    const float* w1    = (const float*)d_in[2];  // [512,16]
    const float* b1    = (const float*)d_in[3];  // [512]
    const float* w2    = (const float*)d_in[4];  // [589824]
    float*       out   = (float*)d_out;          // [32,256,28,28]

    char* ws = (char*)d_ws;
    float*          a_ws    = (float*)ws;                       // A_BYTES
    unsigned*       xpre    = (unsigned*)(ws + A_BYTES);        // X_BYTES
    unsigned short* wpre_hi = (unsigned short*)(ws + A_BYTES + X_BYTES);
    unsigned short* wpre_lo = (unsigned short*)(ws + A_BYTES + X_BYTES + W_BYTES);

    wn_fc1<<<dim3(64), dim3(256), 0, stream>>>(x_gap, w1, b1, a_ws);

    dim3 grid(NSP / BN, OUPC / BM, NBATCH);      // (7, 2, 32)
    if (ws_size >= WS_NEED) {
        wn_xgen<<<dim3(32 * 256 * 784 / 256), dim3(256), 0, stream>>>(x, xpre);
        wn_wgen<<<dim3(32 * 9 * 256 * 256 / 256), dim3(256), 0, stream>>>(
            w2, a_ws, wpre_hi, wpre_lo);
        wn_conv<true><<<grid, dim3(256), 0, stream>>>(
            x, xpre, w2, wpre_hi, wpre_lo, a_ws, out);
    } else {
        wn_conv<false><<<grid, dim3(256), 0, stream>>>(
            x, xpre, w2, wpre_hi, wpre_lo, a_ws, out);
    }
}
